// CasualSelfAttention_20100446945545
// MI455X (gfx1250) — matrix operand order repaired and checked by reading, not compiled, not run on hardware
//
#include <hip/hip_runtime.h>

// Causal self-attention fwd for MI455X (gfx1250, wave32, WMMA).
// Pipeline: [qkv GEMM f32->f16 ws] -> [flash attention, all-WMMA] -> [proj GEMM -> f32 out]
// Workspace layout (f16): Q[2,16,2048,64] | K | V | Y[2,2048,1024]  = 32 MB total.

typedef _Float16 half8 __attribute__((ext_vector_type(8)));
typedef _Float16 v16h  __attribute__((ext_vector_type(16)));
typedef float    v8f   __attribute__((ext_vector_type(8)));

static constexpr int BB  = 2;
static constexpr int T   = 2048;
static constexpr int C   = 1024;
static constexpr int H   = 16;
static constexpr int HD  = 64;
static constexpr int BT  = BB * T;     // 4096
static constexpr int N3C = 3 * C;      // 3072

__device__ __forceinline__ v16h pack16(half8 lo, half8 hi8) {
  union { v16h v; half8 h[2]; } u;
  u.h[0] = lo; u.h[1] = hi8;
  return u.v;
}

__device__ __forceinline__ v8f wmma_f16(v16h a, v16h b, v8f c) {
  // D = A(16x32 f16) * B(32x16 f16) + C(16x16 f32)
  return __builtin_amdgcn_wmma_f32_16x16x32_f16(false, a, false, b, (short)0, c, false, false);
}

// Async global->LDS copy (CDNA5): tracked by ASYNCcnt, no VGPR data movement.
// LDS operand = low 32 bits of the generic shared pointer (flat aperture rule:
// LDS address is the low 32 bits, relative to the wave's LDS base).
__device__ __forceinline__ void async_copy_b128(void* lds_dst, const void* gsrc) {
  asm volatile("global_load_async_to_lds_b128 %0, %1, off"
               :: "v"((unsigned int)(unsigned long long)lds_dst),
                  "v"((unsigned long long)gsrc)
               : "memory");
}
__device__ __forceinline__ void wait_async0() {
  asm volatile("s_wait_asynccnt 0x0" ::: "memory");
}

// ---------------------------------------------------------------------------
// Kernel 1: qkv = x @ w_attn + b_attn ; scatter to Q (pre-scaled), K, V (f16)
// M=4096, N=3072, K=1024. Block tile 128x128, 8 waves (4x2), wave tile 32x64.
// ---------------------------------------------------------------------------
__global__ __launch_bounds__(256) void qkv_kernel(
    const float* __restrict__ x, const float* __restrict__ w,
    const float* __restrict__ bias,
    _Float16* __restrict__ Qg, _Float16* __restrict__ Kg, _Float16* __restrict__ Vg)
{
  __shared__ __align__(32) _Float16 As[128][32];  // row-major   (8 KB)
  __shared__ __align__(32) _Float16 Bs[128][32];  // transposed  (8 KB)
  const int tid = threadIdx.x, lane = tid & 31, wave = tid >> 5;
  const int wr = wave >> 1, wc = wave & 1;        // 4x2 wave grid
  const int ln = lane & 15, hi = lane >> 4;
  const int m0 = blockIdx.y * 128, n0 = blockIdx.x * 128;

  v8f acc[2][4] = {};

  for (int k0 = 0; k0 < C; k0 += 32) {
    if (k0 + 32 < C)  // speculative prefetch of next A tile -> global_prefetch_b8
      __builtin_prefetch(x + (size_t)(m0 + (tid >> 1)) * C + k0 + 32, 0, 0);
    // Stage A: 128x32 f32 -> f16 (1024 float4, 4 per thread)
    #pragma unroll
    for (int i = 0; i < 4; ++i) {
      int idx = tid + i * 256;
      int row = idx >> 3, c4 = (idx & 7) << 2;
      const float4 f = *(const float4*)(x + (size_t)(m0 + row) * C + k0 + c4);
      As[row][c4 + 0] = (_Float16)f.x; As[row][c4 + 1] = (_Float16)f.y;
      As[row][c4 + 2] = (_Float16)f.z; As[row][c4 + 3] = (_Float16)f.w;
    }
    // Stage B transposed: w[k0..+32)[n0..+128) (1024 float4, 4 per thread)
    #pragma unroll
    for (int i = 0; i < 4; ++i) {
      int idx = tid + i * 256;
      int kr = idx >> 5, c4 = (idx & 31) << 2;
      const float4 f = *(const float4*)(w + (size_t)(k0 + kr) * N3C + n0 + c4);
      Bs[c4 + 0][kr] = (_Float16)f.x; Bs[c4 + 1][kr] = (_Float16)f.y;
      Bs[c4 + 2][kr] = (_Float16)f.z; Bs[c4 + 3][kr] = (_Float16)f.w;
    }
    __syncthreads();
    v16h a[2], bfr[4];
    #pragma unroll
    for (int i2 = 0; i2 < 2; ++i2) {
      int row = wr * 32 + i2 * 16 + ln;   // A frag: halves K hi*16..+15
      a[i2] = pack16(*(const half8*)&As[row][hi * 16],
                     *(const half8*)&As[row][hi * 16 + 8]);
    }
    #pragma unroll
    for (int j = 0; j < 4; ++j) {
      int coln = wc * 64 + j * 16 + ln;   // B frag: col n=ln, halves K hi*16..+15
      bfr[j] = *(const v16h*)&Bs[coln][hi * 16];
    }
    #pragma unroll
    for (int i2 = 0; i2 < 2; ++i2)
      #pragma unroll
      for (int j = 0; j < 4; ++j)
        acc[i2][j] = wmma_f16(a[i2], bfr[j], acc[i2][j]);
    __syncthreads();
  }

  // Epilogue: bias, Q pre-scale (1/sqrt(64)=0.125), scatter to [B,H,T,hd] f16
  #pragma unroll
  for (int j = 0; j < 4; ++j) {
    int n = n0 + wc * 64 + j * 16 + ln;
    float bv = bias[n];
    int part = n >> 10, within = n & 1023;
    int h = within >> 6, d = within & 63;
    _Float16* dst = (part == 0) ? Qg : (part == 1) ? Kg : Vg;
    float scl = (part == 0) ? 0.125f : 1.0f;
    #pragma unroll
    for (int i2 = 0; i2 < 2; ++i2)
      #pragma unroll
      for (int r = 0; r < 8; ++r) {
        int m = m0 + wr * 32 + i2 * 16 + r + hi * 8;  // C layout: row = r + 8*hi
        int bidx = m >> 11, tt = m & (T - 1);
        dst[(((size_t)bidx * H + h) * T + tt) * HD + d] =
            (_Float16)((acc[i2][j][r] + bv) * scl);
      }
  }
}

// ---------------------------------------------------------------------------
// Kernel 2: flash attention per (b,h). Block = 8 waves; wave owns 16 q-rows.
// K-blocks of 32 keys; K staged via async global->LDS, V transposed manually.
// ---------------------------------------------------------------------------
__global__ __launch_bounds__(256) void attn_kernel(
    const _Float16* __restrict__ Qg, const _Float16* __restrict__ Kg,
    const _Float16* __restrict__ Vg, _Float16* __restrict__ Yg)
{
  __shared__ __align__(32) _Float16 Ks[32][64];      // 4 KB
  __shared__ __align__(32) _Float16 Vt[64][32];      // 4 KB (transposed)
  __shared__ __align__(32) _Float16 Ps[8][16][32];   // 8 KB per-wave P scratch
  const int tid = threadIdx.x, lane = tid & 31, wave = tid >> 5;
  const int ln = lane & 15, hi = lane >> 4;
  const int b = blockIdx.z, h = blockIdx.y;
  const int q0 = blockIdx.x * 128 + wave * 16;
  const size_t baseBH = ((size_t)b * H + h) * T;
  const _Float16* Qh = Qg + baseBH * HD;
  const _Float16* Kh = Kg + baseBH * HD;
  const _Float16* Vh = Vg + baseBH * HD;

  // Q fragments (16x64 -> two 16x32 A frags), held in registers all kernel
  v16h qf[2];
  {
    const _Float16* qp = Qh + (size_t)(q0 + ln) * HD;
    #pragma unroll
    for (int c = 0; c < 2; ++c)
      qf[c] = pack16(*(const half8*)(qp + c * 32 + hi * 16),
                     *(const half8*)(qp + c * 32 + hi * 16 + 8));
  }

  v8f o[4] = {};                 // 16x64 output accumulator (4 n-tiles)
  float mrow[8], lrow[8];
  #pragma unroll
  for (int r = 0; r < 8; ++r) { mrow[r] = -3.0e38f; lrow[r] = 0.0f; }

  const int kend = blockIdx.x * 128 + 128;   // max keys needed by this block
  const int srow = tid >> 3, sc8 = (tid & 7) << 3;
  for (int kb = 0; kb < kend; kb += 32) {
    __syncthreads();
    // Stage 32 keys: K via async global->LDS (ASYNCcnt), V transposed via VGPRs
    async_copy_b128(&Ks[srow][sc8], Kh + (size_t)(kb + srow) * HD + sc8);
    half8 vv = *(const half8*)(Vh + (size_t)(kb + srow) * HD + sc8);
    #pragma unroll
    for (int j = 0; j < 8; ++j) Vt[sc8 + j][srow] = vv[j];
    wait_async0();
    __syncthreads();
    if (kb > q0 + 15) continue;  // wave-uniform causal skip (EXEC stays all-ones)

    // S = Q K^T : two 16x16 tiles, K-dim 64 = 2 chained WMMAs each
    v8f s[2] = {};
    #pragma unroll
    for (int ct = 0; ct < 2; ++ct)
      #pragma unroll
      for (int c = 0; c < 2; ++c) {
        v16h bk = *(const v16h*)&Ks[ct * 16 + ln][c * 32 + hi * 16];
        s[ct] = wmma_f16(qf[c], bk, s[ct]);
      }
    // Causal mask in C-layout: col = kb+ct*16+ln, row = q0 + r + 8*hi
    #pragma unroll
    for (int ct = 0; ct < 2; ++ct)
      #pragma unroll
      for (int r = 0; r < 8; ++r)
        if (kb + ct * 16 + ln > q0 + r + hi * 8) s[ct][r] = -3.0e38f;

    // Online softmax: rows live across 16-lane halves; xor masks 1..8 stay in-half
    #pragma unroll
    for (int r = 0; r < 8; ++r) {
      float v = fmaxf(s[0][r], s[1][r]);
      #pragma unroll
      for (int off = 1; off < 16; off <<= 1)
        v = fmaxf(v, __shfl_xor(v, off, 32));
      float mn = fmaxf(mrow[r], v);
      float alpha = __expf(mrow[r] - mn);
      mrow[r] = mn;
      float e0 = __expf(s[0][r] - mn), e1 = __expf(s[1][r] - mn);
      s[0][r] = e0; s[1][r] = e1;
      float rs = e0 + e1;
      #pragma unroll
      for (int off = 1; off < 16; off <<= 1)
        rs += __shfl_xor(rs, off, 32);
      lrow[r] = lrow[r] * alpha + rs;
      #pragma unroll
      for (int nt = 0; nt < 4; ++nt) o[nt][r] *= alpha;
    }

    // Re-shape P: C-layout -> A-layout via per-wave LDS scratch
    #pragma unroll
    for (int ct = 0; ct < 2; ++ct)
      #pragma unroll
      for (int r = 0; r < 8; ++r)
        Ps[wave][r + hi * 8][ct * 16 + ln] = (_Float16)s[ct][r];
    v16h pf = pack16(*(const half8*)&Ps[wave][ln][hi * 16],
                     *(const half8*)&Ps[wave][ln][hi * 16 + 8]);

    // O += P @ V : 4 n-tiles of hd, K-dim 32
    #pragma unroll
    for (int nt = 0; nt < 4; ++nt) {
      v16h bv = *(const v16h*)&Vt[nt * 16 + ln][hi * 16];
      o[nt] = wmma_f16(pf, bv, o[nt]);
    }
  }

  // Normalize and write Y as f16 [B,T,C] (C index = h*64 + d)
  #pragma unroll
  for (int r = 0; r < 8; ++r) {
    float inv = 1.0f / lrow[r];
    int t = q0 + r + hi * 8;
    size_t rowbase = ((size_t)b * T + t) * C + h * HD;
    #pragma unroll
    for (int nt = 0; nt < 4; ++nt)
      Yg[rowbase + nt * 16 + ln] = (_Float16)(o[nt][r] * inv);
  }
}

// ---------------------------------------------------------------------------
// Kernel 3: out = Y @ w_proj + b_proj. M=4096, N=1024, K=1024, f32 out.
// Block tile 128x128, wave tile 32x64. A staged via async global->LDS.
// ---------------------------------------------------------------------------
__global__ __launch_bounds__(256) void proj_kernel(
    const _Float16* __restrict__ Y, const float* __restrict__ w,
    const float* __restrict__ bias, float* __restrict__ out)
{
  __shared__ __align__(32) _Float16 As[128][32];
  __shared__ __align__(32) _Float16 Bs[128][32];
  const int tid = threadIdx.x, lane = tid & 31, wave = tid >> 5;
  const int wr = wave >> 1, wc = wave & 1;
  const int ln = lane & 15, hi = lane >> 4;
  const int m0 = blockIdx.y * 128, n0 = blockIdx.x * 128;

  v8f acc[2][4] = {};

  for (int k0 = 0; k0 < C; k0 += 32) {
    if (k0 + 32 < C)
      __builtin_prefetch(w + (size_t)(k0 + 32 + (tid >> 4)) * C + n0, 0, 0);
    // A: pure f16 copy -> async global->LDS (512 b128, 2 per thread)
    #pragma unroll
    for (int i = 0; i < 2; ++i) {
      int idx = tid + i * 256;
      int row = idx >> 2, c8 = (idx & 3) << 3;
      async_copy_b128(&As[row][c8], Y + (size_t)(m0 + row) * C + k0 + c8);
    }
    // B transposed with f32->f16 conversion (1024 float4, 4 per thread)
    #pragma unroll
    for (int i = 0; i < 4; ++i) {
      int idx = tid + i * 256;
      int kr = idx >> 5, c4 = (idx & 31) << 2;
      const float4 f = *(const float4*)(w + (size_t)(k0 + kr) * C + n0 + c4);
      Bs[c4 + 0][kr] = (_Float16)f.x; Bs[c4 + 1][kr] = (_Float16)f.y;
      Bs[c4 + 2][kr] = (_Float16)f.z; Bs[c4 + 3][kr] = (_Float16)f.w;
    }
    wait_async0();
    __syncthreads();
    v16h a[2], bfr[4];
    #pragma unroll
    for (int i2 = 0; i2 < 2; ++i2) {
      int row = wr * 32 + i2 * 16 + ln;
      a[i2] = pack16(*(const half8*)&As[row][hi * 16],
                     *(const half8*)&As[row][hi * 16 + 8]);
    }
    #pragma unroll
    for (int j = 0; j < 4; ++j) {
      int coln = wc * 64 + j * 16 + ln;
      bfr[j] = *(const v16h*)&Bs[coln][hi * 16];
    }
    #pragma unroll
    for (int i2 = 0; i2 < 2; ++i2)
      #pragma unroll
      for (int j = 0; j < 4; ++j)
        acc[i2][j] = wmma_f16(a[i2], bfr[j], acc[i2][j]);
    __syncthreads();
  }

  #pragma unroll
  for (int j = 0; j < 4; ++j) {
    int n = n0 + wc * 64 + j * 16 + ln;
    float bv = bias[n];
    #pragma unroll
    for (int i2 = 0; i2 < 2; ++i2)
      #pragma unroll
      for (int r = 0; r < 8; ++r) {
        int m = m0 + wr * 32 + i2 * 16 + r + hi * 8;
        out[(size_t)m * C + n] = acc[i2][j][r] + bv;
      }
  }
}

// ---------------------------------------------------------------------------
extern "C" void kernel_launch(void* const* d_in, const int* in_sizes, int n_in,
                              void* d_out, int out_size, void* d_ws, size_t ws_size,
                              hipStream_t stream) {
  const float* x      = (const float*)d_in[0];
  const float* w_attn = (const float*)d_in[1];
  const float* b_attn = (const float*)d_in[2];
  const float* w_proj = (const float*)d_in[3];
  const float* b_proj = (const float*)d_in[4];
  float* out = (float*)d_out;

  const size_t QKVN = (size_t)BB * H * T * HD;   // 4M elems each
  _Float16* Qg = (_Float16*)d_ws;
  _Float16* Kg = Qg + QKVN;
  _Float16* Vg = Kg + QKVN;
  _Float16* Yg = Vg + QKVN;                      // 4M elems; 32 MB ws total

  qkv_kernel<<<dim3(N3C / 128, BT / 128), 256, 0, stream>>>(x, w_attn, b_attn, Qg, Kg, Vg);
  attn_kernel<<<dim3(T / 128, H, BB), 256, 0, stream>>>(Qg, Kg, Vg, Yg);
  proj_kernel<<<dim3(C / 128, BT / 128), 256, 0, stream>>>(Yg, w_proj, b_proj, out);
}
